// AlbertDecoderAttention_6253472383101
// MI455X (gfx1250) — matrix operand order repaired and checked by reading, not compiled, not run on hardware
//
#include <hip/hip_runtime.h>
#include <hip/hip_bf16.h>

// ---------------------------------------------------------------------------
// Problem constants (AlbertDecoderAttention reference)
// ---------------------------------------------------------------------------
constexpr int Bn = 4;
constexpr int Tn = 1024;   // decoder length
constexpr int Sn = 1024;   // encoder length
constexpr int Hn = 1024;   // hidden
constexpr int Nh = 16;     // heads
constexpr int Dh = 64;     // head dim
constexpr int Mrows = Bn * Tn;  // 4096

typedef __attribute__((ext_vector_type(16))) __bf16 v16bf;
typedef __attribute__((ext_vector_type(8)))  float  v8f;
typedef __attribute__((ext_vector_type(4)))  unsigned int u32x4;
typedef __attribute__((ext_vector_type(8)))  unsigned int u32x8;

#define WMMA_BF16(a, b, c) \
  __builtin_amdgcn_wmma_f32_16x16x32_bf16(false, (a), false, (b), (short)0, (c), false, false)

// ---------------------------------------------------------------------------
// Fragment loaders per CDNA5 ISA 16-bit WMMA VGPR layouts (wave32).
// With 16B-aligned tiles and ld % 8 == 0, each lane's reads are one or two
// contiguous, 16B-aligned 8x-bf16 runs -> compiler emits ds_load_b128.
// ---------------------------------------------------------------------------
__device__ __forceinline__ v16bf frag_a16x32(const __bf16* t, int ld, int lane) {
  int row = lane & 15, hf = lane >> 4;
  v16bf f;
#pragma unroll
  for (int v = 0; v < 8; ++v) {
    int k = (v < 4) ? (hf * 8 + 2 * v) : (16 + hf * 8 + 2 * (v - 4));
    f[2 * v]     = t[row * ld + k];
    f[2 * v + 1] = t[row * ld + k + 1];
  }
  return f;
}

// B[k][n] from tile stored row-major as t[k][n] (strided pairs; used for V)
__device__ __forceinline__ v16bf frag_b32x16(const __bf16* t, int ld, int lane) {
  int n = lane & 15, hf = lane >> 4;
  v16bf f;
#pragma unroll
  for (int v = 0; v < 8; ++v) {
    int k = (v < 4) ? (hf * 8 + 2 * v) : (16 + hf * 8 + 2 * (v - 4));
    f[2 * v]     = t[k * ld + n];
    f[2 * v + 1] = t[(k + 1) * ld + n];
  }
  return f;
}

// B[k][n] where the tile is stored transposed: t[n][k]; two contiguous
// 8-elt (16B) runs per lane -> 2x ds_load_b128.
__device__ __forceinline__ v16bf frag_bT(const __bf16* t, int ld, int lane) {
  int n = lane & 15, hf = lane >> 4;
  v16bf f;
#pragma unroll
  for (int v = 0; v < 8; ++v) {
    int k = (v < 4) ? (hf * 8 + 2 * v) : (16 + hf * 8 + 2 * (v - 4));
    f[2 * v]     = t[n * ld + k];
    f[2 * v + 1] = t[n * ld + k + 1];
  }
  return f;
}

// ---------------------------------------------------------------------------
// Tensor Data Mover: 2D bf16 tile (tile_d0 elems/row x tile_rows rows) from
// global (row stride = stride_elems) into LDS at lds_off, with LDS row
// padding so the staged row stride becomes tile_d0 + 8 bf16 (pad 4 dwords
// after every 32 dwords = one 64-elt row).  Hand-packed D# per CDNA5 ISA §8.
// ---------------------------------------------------------------------------
__device__ __forceinline__ void tdm_load_tile_bf16(
    unsigned lds_off, const __bf16* gsrc,
    unsigned tensor_d0, unsigned tensor_rows,
    unsigned tile_d0, unsigned tile_rows, unsigned stride_elems) {
  unsigned long long ga = (unsigned long long)(size_t)gsrc;
  u32x4 g0;
  g0[0] = 1u;                                    // count=1, user descriptor
  g0[1] = lds_off;                               // lds_addr (bytes)
  g0[2] = (unsigned)(ga & 0xFFFFFFFFu);          // global_addr[31:0]
  g0[3] = (unsigned)((ga >> 32) & 0x01FFFFFFu) | 0x80000000u;  // addr[56:32] | type=2
  u32x8 g1;
  g1[0] = (1u << 16)      // data_size = 1 (2 bytes)
        | (1u << 20)      // pad_enable
        | (4u << 22)      // pad_interval: code 4 = every 32 dwords (one 64-elt row)
        | (3u << 25);     // pad_amount:  code 3 = 4 dwords (8 bf16) -> row stride 72
  g1[1] = (tensor_d0 & 0xFFFFu) << 16;                         // tensor_dim0[15:0]
  g1[2] = (tensor_d0 >> 16) | ((tensor_rows & 0xFFFFu) << 16); // d0 hi | dim1 lo
  g1[3] = (tensor_rows >> 16) | (tile_d0 << 16);               // dim1 hi | tile_dim0
  g1[4] = tile_rows;                                           // tile_dim1 (tile_dim2=0)
  g1[5] = stride_elems;                                        // tensor_dim0_stride[31:0]
  g1[6] = 0u;
  g1[7] = 0u;
  asm volatile("tensor_load_to_lds %0, %1" :: "s"(g0), "s"(g1) : "memory");
}

__device__ __forceinline__ void tdm_wait0() {
  asm volatile("s_wait_tensorcnt 0x0" ::: "memory");
}

// ---------------------------------------------------------------------------
// GEMM: Y[M,N] = X[M,K] @ W[N,K]^T + bias, fp32 in/out, bf16 WMMA inside.
// Optional bf16 side-copy of Y (for downstream attention consumers).
// Block: 256 threads (8 waves), tile 128x128, K step 32; wave = 32x64.
// ---------------------------------------------------------------------------
constexpr int GLD = 40;  // LDS stride: 80B rows -> 16B aligned + conflict-free

__global__ __launch_bounds__(256) void gemm_bf16_wmma(
    const float* __restrict__ X, const float* __restrict__ W,
    const float* __restrict__ bias, float* __restrict__ Y,
    __bf16* __restrict__ Ybf, int M, int K, int Nout) {
  __shared__ alignas(16) __bf16 At[128 * GLD];
  __shared__ alignas(16) __bf16 Bt[128 * GLD];

  const int tid = threadIdx.x;
  const int lane = tid & 31, wid = tid >> 5;
  const int wm = wid >> 1, wn = wid & 1;
  const int M0 = blockIdx.y * 128, N0 = blockIdx.x * 128;

  v8f acc[2][4];
#pragma unroll
  for (int i = 0; i < 2; ++i)
#pragma unroll
    for (int j = 0; j < 4; ++j)
#pragma unroll
      for (int e = 0; e < 8; ++e) acc[i][j][e] = 0.0f;

  const int lr = tid >> 1;         // 0..127: tile row to stage
  const int lc = (tid & 1) * 16;   // 16-element column segment

  for (int k0 = 0; k0 < K; k0 += 32) {
    const float* xs = X + (size_t)(M0 + lr) * K + k0 + lc;
    const float* ws = W + (size_t)(N0 + lr) * K + k0 + lc;
    // prefetch next K-step rows (global_prefetch_b8; speculative tail is OK)
    __builtin_prefetch(xs + 32, 0, 3);
    __builtin_prefetch(ws + 32, 0, 3);
#pragma unroll
    for (int i = 0; i < 16; ++i) {
      At[lr * GLD + lc + i] = (__bf16)xs[i];
      Bt[lr * GLD + lc + i] = (__bf16)ws[i];
    }
    __syncthreads();

    v16bf a0 = frag_a16x32(At + (wm * 32) * GLD, GLD, lane);
    v16bf a1 = frag_a16x32(At + (wm * 32 + 16) * GLD, GLD, lane);
    v16bf bf[4];
#pragma unroll
    for (int nf = 0; nf < 4; ++nf)
      bf[nf] = frag_bT(Bt + (wn * 64 + nf * 16) * GLD, GLD, lane);
#pragma unroll
    for (int nf = 0; nf < 4; ++nf) {
      acc[0][nf] = WMMA_BF16(a0, bf[nf], acc[0][nf]);
      acc[1][nf] = WMMA_BF16(a1, bf[nf], acc[1][nf]);
    }
    __syncthreads();
  }

  // Epilogue: hoist the (uniform) bf16 side-copy test out of the store loops
  // so each variant is a straight-line store clause.
  const int cb = lane & 15, hf = lane >> 4;
  if (Ybf != nullptr) {
#pragma unroll
    for (int mf = 0; mf < 2; ++mf)
#pragma unroll
      for (int nf = 0; nf < 4; ++nf) {
        int col = N0 + wn * 64 + nf * 16 + cb;
        float bv = bias[col];
#pragma unroll
        for (int r = 0; r < 8; ++r) {
          int row = M0 + wm * 32 + mf * 16 + hf * 8 + r;
          float y = acc[mf][nf][r] + bv;
          Y[(size_t)row * Nout + col] = y;
          Ybf[(size_t)row * Nout + col] = (__bf16)y;
        }
      }
  } else {
#pragma unroll
    for (int mf = 0; mf < 2; ++mf)
#pragma unroll
      for (int nf = 0; nf < 4; ++nf) {
        int col = N0 + wn * 64 + nf * 16 + cb;
        float bv = bias[col];
#pragma unroll
        for (int r = 0; r < 8; ++r) {
          int row = M0 + wm * 32 + mf * 16 + hf * 8 + r;
          Y[(size_t)row * Nout + col] = acc[mf][nf][r] + bv;
        }
      }
  }
}

// ---------------------------------------------------------------------------
// Flash attention (no 1/sqrt(d) scaling, matching reference), bf16 WMMA.
// Q/K/V consumed from bf16 copies; K/V tiles staged by the TDM.
// Grid: (Tq/64, Nh, Bn); block 128 threads = 4 waves; wave owns 16 q rows.
// mask3d!=0: mask indexed [b][q][k] (tgt mask); else [b][k] (src mask).
// ---------------------------------------------------------------------------
constexpr int KLD = 72;  // TDM-padded row stride: 64 data + 8 pad bf16
constexpr int PLD = 40;  // P scratch stride (16B-aligned rows)

__global__ __launch_bounds__(128) void flash_attn_wmma(
    const __bf16* __restrict__ Qbf, const __bf16* __restrict__ Kbf,
    const __bf16* __restrict__ Vbf, const int* __restrict__ mask,
    int mask3d, float* __restrict__ Ctx, int Tq, int Tk) {
  __shared__ alignas(16) __bf16 Kt[32 * KLD];
  __shared__ alignas(16) __bf16 Vt[32 * KLD];
  __shared__ alignas(16) __bf16 Pt[4 * 16 * PLD];

  const int tid = threadIdx.x;
  const int lane = tid & 31, wid = tid >> 5;
  const int col = lane & 15, hf = lane >> 4;
  const int head = blockIdx.y, b = blockIdx.z;
  const int hoff = head * Dh;
  const int qrow0 = blockIdx.x * 64 + wid * 16;  // first q row of this wave

  // Q fragments (16 rows x 64) in registers; bf16 global, 16B-aligned runs.
  const __bf16* qb = Qbf + ((size_t)b * Tq + qrow0) * Hn + hoff;
  v16bf qf0, qf1;
#pragma unroll
  for (int v = 0; v < 8; ++v) {
    int k = (v < 4) ? (hf * 8 + 2 * v) : (16 + hf * 8 + 2 * (v - 4));
    qf0[2 * v]     = qb[col * Hn + k];
    qf0[2 * v + 1] = qb[col * Hn + k + 1];
    qf1[2 * v]     = qb[col * Hn + 32 + k];
    qf1[2 * v + 1] = qb[col * Hn + 32 + k + 1];
  }

  float mst[8], lst[8];
  v8f acc[4];
#pragma unroll
  for (int r = 0; r < 8; ++r) { mst[r] = -3.0e38f; lst[r] = 0.0f; }
#pragma unroll
  for (int j = 0; j < 4; ++j)
#pragma unroll
    for (int e = 0; e < 8; ++e) acc[j][e] = 0.0f;

  __bf16* Pw = Pt + wid * 16 * PLD;
  const unsigned kt_off = (unsigned)(size_t)(void*)&Kt[0];
  const unsigned vt_off = (unsigned)(size_t)(void*)&Vt[0];

  for (int kb = 0; kb < Tk; kb += 32) {
    // TDM stages K/V 32x64 bf16 tiles (pad -> row stride 72); wave 0 issues.
    if (wid == 0) {
      const __bf16* ksrc = Kbf + ((size_t)b * Tk + kb) * Hn + hoff;
      const __bf16* vsrc = Vbf + ((size_t)b * Tk + kb) * Hn + hoff;
      tdm_load_tile_bf16(kt_off, ksrc, Hn, (unsigned)(Bn * Tk), Dh, 32, Hn);
      tdm_load_tile_bf16(vt_off, vsrc, Hn, (unsigned)(Bn * Tk), Dh, 32, Hn);
      tdm_wait0();
    }
    __syncthreads();

    // S = Q @ K^T : two 16x16 tiles (keys kb..+15 and kb+16..+31)
    v8f z;
#pragma unroll
    for (int e = 0; e < 8; ++e) z[e] = 0.0f;

    v8f s0 = WMMA_BF16(qf0, frag_bT(Kt, KLD, lane), z);
    s0     = WMMA_BF16(qf1, frag_bT(Kt + 32, KLD, lane), s0);
    v8f s1 = WMMA_BF16(qf0, frag_bT(Kt + 16 * KLD, KLD, lane), z);
    s1     = WMMA_BF16(qf1, frag_bT(Kt + 16 * KLD + 32, KLD, lane), s1);

    // mask + online softmax (row state duplicated across each 16-lane half)
#pragma unroll
    for (int r = 0; r < 8; ++r) {
      int qg = qrow0 + hf * 8 + r;
      int k0g = kb + col, k1g = kb + 16 + col;
      int m0, m1;
      if (mask3d) {
        const int* mrow = mask + ((size_t)b * Tq + qg) * Tk;
        m0 = mrow[k0g]; m1 = mrow[k1g];
      } else {
        m0 = mask[(size_t)b * Tk + k0g]; m1 = mask[(size_t)b * Tk + k1g];
      }
      float s0v = m0 ? s0[r] : -1.0e6f;
      float s1v = m1 ? s1[r] : -1.0e6f;

      float mx = fmaxf(s0v, s1v);
#pragma unroll
      for (int off = 8; off; off >>= 1) mx = fmaxf(mx, __shfl_xor(mx, off, 16));
      float mnew = fmaxf(mst[r], mx);
      float p0 = __expf(s0v - mnew);
      float p1 = __expf(s1v - mnew);
      float rs = p0 + p1;
#pragma unroll
      for (int off = 8; off; off >>= 1) rs += __shfl_xor(rs, off, 16);
      float corr = __expf(mst[r] - mnew);
      mst[r] = mnew;
      lst[r] = lst[r] * corr + rs;
      acc[0][r] *= corr; acc[1][r] *= corr; acc[2][r] *= corr; acc[3][r] *= corr;

      int prow = hf * 8 + r;
      Pw[prow * PLD + col]      = (__bf16)p0;
      Pw[prow * PLD + 16 + col] = (__bf16)p1;
    }

    // ctx += P @ V  (P as A-fragment via per-wave LDS scratch)
    v16bf pf = frag_a16x32(Pw, PLD, lane);
#pragma unroll
    for (int j = 0; j < 4; ++j) {
      v16bf vf = frag_b32x16(Vt + j * 16, KLD, lane);
      acc[j] = WMMA_BF16(pf, vf, acc[j]);
    }
    __syncthreads();
  }

  const size_t orow0 = (size_t)b * Tq + qrow0;
#pragma unroll
  for (int j = 0; j < 4; ++j)
#pragma unroll
    for (int r = 0; r < 8; ++r) {
      float inv = 1.0f / lst[r];
      Ctx[(orow0 + hf * 8 + r) * Hn + hoff + j * 16 + col] = acc[j][r] * inv;
    }
}

// ---------------------------------------------------------------------------
// Fused residual add + LayerNorm: out = LN(resid + proj) * g + b
// One 256-thread block per row of 1024.
// ---------------------------------------------------------------------------
__global__ __launch_bounds__(256) void ln_residual(
    const float* __restrict__ resid, const float* __restrict__ proj,
    const float* __restrict__ g, const float* __restrict__ beta,
    float* __restrict__ out) {
  __shared__ float red[256];
  const int row = blockIdx.x, t = threadIdx.x;
  const float* r0 = resid + (size_t)row * Hn;
  const float* p0 = proj + (size_t)row * Hn;

  float x[4];
  float s = 0.0f;
#pragma unroll
  for (int i = 0; i < 4; ++i) {
    x[i] = r0[t + i * 256] + p0[t + i * 256];
    s += x[i];
  }
  red[t] = s;
  __syncthreads();
  for (int o = 128; o; o >>= 1) {
    if (t < o) red[t] += red[t + o];
    __syncthreads();
  }
  float mu = red[0] * (1.0f / Hn);
  __syncthreads();

  float v = 0.0f;
#pragma unroll
  for (int i = 0; i < 4; ++i) {
    float d = x[i] - mu;
    v += d * d;
  }
  red[t] = v;
  __syncthreads();
  for (int o = 128; o; o >>= 1) {
    if (t < o) red[t] += red[t + o];
    __syncthreads();
  }
  float inv = rsqrtf(red[0] * (1.0f / Hn) + 1e-12f);

#pragma unroll
  for (int i = 0; i < 4; ++i) {
    int c = t + i * 256;
    out[(size_t)row * Hn + c] = (x[i] - mu) * inv * g[c] + beta[c];
  }
}

// ---------------------------------------------------------------------------
// Orchestration
// ---------------------------------------------------------------------------
extern "C" void kernel_launch(void* const* d_in, const int* in_sizes, int n_in,
                              void* d_out, int out_size, void* d_ws, size_t ws_size,
                              hipStream_t stream) {
  const float* enc   = (const float*)d_in[0];
  const float* dec   = (const float*)d_in[1];
  const int*   smask = (const int*)d_in[2];
  const int*   tmask = (const int*)d_in[3];
  const float* q_w = (const float*)d_in[4],  * q_b = (const float*)d_in[5];
  const float* k_w = (const float*)d_in[6],  * k_b = (const float*)d_in[7];
  const float* v_w = (const float*)d_in[8],  * v_b = (const float*)d_in[9];
  const float* sq_w = (const float*)d_in[10], * sq_b = (const float*)d_in[11];
  const float* sk_w = (const float*)d_in[12], * sk_b = (const float*)d_in[13];
  const float* sv_w = (const float*)d_in[14], * sv_b = (const float*)d_in[15];
  const float* dw = (const float*)d_in[16], * db = (const float*)d_in[17];
  const float* lg = (const float*)d_in[18], * lb = (const float*)d_in[19];

  float* ws = (float*)d_ws;
  const size_t SZ = (size_t)Mrows * Hn;  // 4M elements
  float* Aq = ws + 0 * SZ;  // q / q2 (also residual)
  float* Bk = ws + 1 * SZ;  // k / k2
  float* Cv = ws + 2 * SZ;  // v / v2
  float* Dc = ws + 3 * SZ;  // ctx
  float* Ep = ws + 4 * SZ;  // dense proj
  float* Fs = ws + 5 * SZ;  // self_out
  __bf16* bfb = (__bf16*)(ws + 6 * SZ);
  __bf16* Qbf = bfb + 0 * SZ;
  __bf16* Kbf = bfb + 1 * SZ;
  __bf16* Vbf = bfb + 2 * SZ;

  dim3 gGrid(Hn / 128, Mrows / 128);  // (8, 32)
  dim3 gBlk(256);
  dim3 fGrid(Tn / 64, Nh, Bn);        // (16, 16, 4)
  dim3 fBlk(128);
  dim3 lGrid(Mrows);
  dim3 lBlk(256);

  // ---- self attention ----
  gemm_bf16_wmma<<<gGrid, gBlk, 0, stream>>>(dec, q_w, q_b, Aq, Qbf, Mrows, Hn, Hn);
  gemm_bf16_wmma<<<gGrid, gBlk, 0, stream>>>(dec, k_w, k_b, Bk, Kbf, Mrows, Hn, Hn);
  gemm_bf16_wmma<<<gGrid, gBlk, 0, stream>>>(dec, v_w, v_b, Cv, Vbf, Mrows, Hn, Hn);
  flash_attn_wmma<<<fGrid, fBlk, 0, stream>>>(Qbf, Kbf, Vbf, tmask, 1, Dc, Tn, Tn);
  gemm_bf16_wmma<<<gGrid, gBlk, 0, stream>>>(Dc, dw, db, Ep, nullptr, Mrows, Hn, Hn);
  ln_residual<<<lGrid, lBlk, 0, stream>>>(Aq, Ep, lg, lb, Fs);

  // ---- cross attention ----
  gemm_bf16_wmma<<<gGrid, gBlk, 0, stream>>>(Fs, sq_w, sq_b, Aq, Qbf, Mrows, Hn, Hn);
  gemm_bf16_wmma<<<gGrid, gBlk, 0, stream>>>(enc, sk_w, sk_b, Bk, Kbf, Mrows, Hn, Hn);
  gemm_bf16_wmma<<<gGrid, gBlk, 0, stream>>>(enc, sv_w, sv_b, Cv, Vbf, Mrows, Hn, Hn);
  flash_attn_wmma<<<fGrid, fBlk, 0, stream>>>(Qbf, Kbf, Vbf, smask, 0, Dc, Tn, Sn);
  gemm_bf16_wmma<<<gGrid, gBlk, 0, stream>>>(Dc, dw, db, Ep, nullptr, Mrows, Hn, Hn);
  ln_residual<<<lGrid, lBlk, 0, stream>>>(Aq, Ep, lg, lb, (float*)d_out);
}
